// AutoAttention_35768487641206
// MI455X (gfx1250) — matrix operand order repaired and checked by reading, not compiled, not run on hardware
//
#include <hip/hip_runtime.h>

// ---------------------------------------------------------------------------
// Fused "AutoAttention" for MI455X (gfx1250, wave32, WMMA + Tensor Data Mover).
//
// Algebraic reduction: with Xroll_s[h] = X[(h-sP) mod H],
//   Q   = X @ Wq^T + bq
//   G   = Q @ Wk                      (scores_s = sum_h G[h]*X[(h-sP)%H])
//   att = softmax(scores / sqrt(H))   (bk cancels inside softmax)
//   y[h]= sum_s att_s * X[(h-sP)%H]
//   ctx = y @ Wv^T + bv               (sum att = 1 keeps bv exact)
//   out = ctx @ Wo^T + bo
// => 4 GEMMs [32768,512]x[512,512] fused per 64-row tile, bf16 WMMA fp32 acc.
// Weights are staged per-wave into LDS by TDM (tensor_load_to_lds), double
// buffered on TENSORcnt, so the WMMA loop reads only LDS.
// ---------------------------------------------------------------------------

typedef __bf16 bf16_t;
typedef __bf16 v8bf  __attribute__((ext_vector_type(8)));
typedef __bf16 v16bf __attribute__((ext_vector_type(16)));
typedef float  v8f   __attribute__((ext_vector_type(8)));
typedef unsigned int u32x4 __attribute__((ext_vector_type(4)));
typedef int          i32x4 __attribute__((ext_vector_type(4)));
typedef int          i32x8 __attribute__((ext_vector_type(8)));

#define HDIM     512
#define PSHIFT   64
#define NSHIFT   8
#define MTILE    64
#define RS       520                 // LDS row stride (bf16 elems) for X/Q/G tiles
#define NTHREADS 256
#define NROWS_TOTAL (16 * 2048)      // B * N

#define TILE_BYTES   (MTILE * RS * 2)          // 66,560 per tile buffer
#define SLAB_ROWB    80                        // 64B row + 16B TDM pad -> no bank conflicts
#define SLAB_BYTES   (64 * SLAB_ROWB)          // 5,120: one 64-row x 32-k weight slab
#define SLAB_BASE    (3 * TILE_BYTES)          // slabs start after Xs/Qs/Gs
#define SMEM_TOTAL   (SLAB_BASE + 8 * 2 * SLAB_BYTES)   // 281,600 B (< 320KB WGP LDS)

union BF16x16 { v16bf v; v8bf h[2]; };

// A-matrix 16x32 bf16 fragment per ISA 7.12.2:
// lane l: row = l&15, hi = l>>4; halfs 0..7 = K(k0+hi*8..), 8..15 = K(k0+16+hi*8..)
__device__ __forceinline__ v16bf load_a_frag(const bf16_t* __restrict__ row,
                                             int k0, int hi) {
  BF16x16 u;
  u.h[0] = *(const v8bf*)(row + k0 + hi * 8);
  u.h[1] = *(const v8bf*)(row + k0 + hi * 8 + 16);
  return u.v;
}

// B-matrix 32x16 bf16 fragment from a TDM-staged LDS slab:
// lane l holds W_b row (n_local), halfs 0..7 = K(hi*8..), 8..15 = K(16+hi*8..).
// Slab row stride 80B: addresses 16B-aligned, 20*lane%64 distinct -> conflict-free.
__device__ __forceinline__ v16bf load_b_frag_lds(const char* __restrict__ slab,
                                                 int n_local, int hi) {
  const bf16_t* p = (const bf16_t*)(slab + n_local * SLAB_ROWB + hi * 16);
  BF16x16 u;
  u.h[0] = *(const v8bf*)(p);
  u.h[1] = *(const v8bf*)(p + 16);
  return u.v;
}

// Issue one TDM load: weight rows [colbase, colbase+64) x K [kk*32, kk*32+32)
// (bf16, tensor row stride 512) -> LDS slab at byte offset lds_off.
// D# per ISA ch.8: group0 = {count=1 | lds_addr | global_addr | type=2},
// group1 = {data_size=2B, pad: every 16 DW insert 4 DW, dims/strides}, 2D tile.
// Toolchain builtin is the 6-arg clang-23 form: groups 0..3 + zero i32x8 + cpol.
__device__ __forceinline__ void tdm_issue(const bf16_t* __restrict__ W,
                                          int colbase, int kk, unsigned lds_off) {
  const unsigned long long ga =
      (unsigned long long)(const void*)(W + ((size_t)colbase * HDIM + (size_t)kk * 32));
  u32x4 g0;
  g0.x = 1u;                                     // count=1 (valid user D#)
  g0.y = lds_off;                                // LDS byte address of tile
  g0.z = (unsigned)ga;                           // global_addr[31:0]
  g0.w = ((unsigned)(ga >> 32) & 0x01FFFFFFu)    // global_addr[56:32]
         | 0x80000000u;                          // type=2 ("image")
  i32x8 g1;
  g1[0] = (int)((1u << 16)      // data_size: 2-byte elements
                | (1u << 20)    // pad_enable
                | (3u << 22)    // pad_interval: 16 DWORDs (= one 32-elem row)
                | (3u << 25));  // pad_amount: 4 DWORDs -> 80B LDS row stride
  g1[1] = (int)(512u << 16);    // tensor_dim0 = 512 (bits 79:48, low half)
  g1[2] = (int)(512u << 16);    // tensor_dim0 hi=0 | tensor_dim1 = 512 (low half)
  g1[3] = (int)(32u << 16);     // tensor_dim1 hi=0 | tile_dim0 = 32
  g1[4] = 64;                   // tile_dim1 = 64 rows | tile_dim2 = 0
  g1[5] = 512;                  // tensor_dim0_stride[31:0] = 512 elements
  g1[6] = 0;                    // stride0 hi | tensor_dim1_stride lo (unused, 2D)
  g1[7] = 0;
  const i32x4 gz4 = {0, 0, 0, 0};                // groups 2/3: dims >2 unused
  const i32x8 gz8 = {0, 0, 0, 0, 0, 0, 0, 0};    // extra arg (clang-23): zero-filled
  __builtin_amdgcn_tensor_load_to_lds(g0, g1, gz4, gz4, gz8, 0);
}

// Y[MTILE][512] = A[MTILE][512] @ W_b^T (+bias). A in LDS bf16 (stride RS),
// W_b bf16 [512][512] row-major, global (L2-hot), staged via TDM. Each wave:
// its own 64-col strip, all 4 row-blocks -> 4x4 accum tiles, 16 k-steps of 32.
__device__ __forceinline__ void gemm_tile(
    const bf16_t* __restrict__ As,
    const bf16_t* __restrict__ W,
    char* __restrict__ slab0,         // this wave's 2 staging buffers (LDS)
    unsigned slab0_off,               // LDS byte offset of slab0
    bf16_t* __restrict__ Os,          // LDS bf16 out (or nullptr)
    float* __restrict__ Og,           // global fp32 out, stride HDIM (or nullptr)
    const float* __restrict__ bias,   // fp32 [512] (or nullptr)
    int wave, int lane)
{
  const int lo = lane & 15, hi = lane >> 4;
  const int colbase = wave * 64;

  v8f acc[4][4];
#pragma unroll
  for (int rb = 0; rb < 4; ++rb)
#pragma unroll
    for (int ct = 0; ct < 4; ++ct)
      acc[rb][ct] = (v8f){0.f, 0.f, 0.f, 0.f, 0.f, 0.f, 0.f, 0.f};

  tdm_issue(W, colbase, 0, slab0_off);           // prefetch k-step 0 -> buf 0

  for (int kk = 0; kk < HDIM / 32; ++kk) {
    if (kk + 1 < HDIM / 32) {                    // prefetch next slab -> other buf
      tdm_issue(W, colbase, kk + 1, slab0_off + ((unsigned)((kk + 1) & 1)) * SLAB_BYTES);
      __builtin_amdgcn_s_wait_tensorcnt(1);      // slab kk landed; kk+1 in flight
    } else {
      __builtin_amdgcn_s_wait_tensorcnt(0);
    }
    asm volatile("" ::: "memory");               // keep LDS reads below the wait

    const char* slab = slab0 + (kk & 1) * SLAB_BYTES;
    const int k0 = kk * 32;
    v16bf bfr[4], afr[4];
#pragma unroll
    for (int ct = 0; ct < 4; ++ct)
      bfr[ct] = load_b_frag_lds(slab, ct * 16 + lo, hi);
#pragma unroll
    for (int rb = 0; rb < 4; ++rb)
      afr[rb] = load_a_frag(As + (rb * 16 + lo) * RS, k0, hi);
#pragma unroll
    for (int rb = 0; rb < 4; ++rb)
#pragma unroll
      for (int ct = 0; ct < 4; ++ct)
        acc[rb][ct] = __builtin_amdgcn_wmma_f32_16x16x32_bf16(
            false, afr[rb], false, bfr[ct], (short)0, acc[rb][ct], false, false);
  }

  // C/D layout: VGPR j, lane l -> M = j + 8*(l>>4), N = l&15 (ISA 7.12.2)
#pragma unroll
  for (int rb = 0; rb < 4; ++rb) {
#pragma unroll
    for (int ct = 0; ct < 4; ++ct) {
      const int col = colbase + ct * 16 + lo;
      const float bb = bias ? bias[col] : 0.f;
#pragma unroll
      for (int j = 0; j < 8; ++j) {
        const int m = rb * 16 + hi * 8 + j;
        const float v = acc[rb][ct][j] + bb;
        if (Os) Os[m * RS + col] = (bf16_t)v;
        else    Og[(size_t)m * HDIM + col] = v;
      }
    }
  }
}

// ---------------------------------------------------------------------------
// Prep: bf16 weight copies in workspace; Wk stored transposed so G = Q @ Wk
// reuses the common Y = A @ W_b^T routine.
// ---------------------------------------------------------------------------
__global__ __launch_bounds__(256) void prep_weights(
    const float* __restrict__ Wq, const float* __restrict__ Wk,
    const float* __restrict__ Wv, const float* __restrict__ Wo,
    bf16_t* __restrict__ wq, bf16_t* __restrict__ wkT,
    bf16_t* __restrict__ wv, bf16_t* __restrict__ wo)
{
  for (int i = blockIdx.x * blockDim.x + threadIdx.x; i < HDIM * HDIM;
       i += gridDim.x * blockDim.x) {
    wq[i] = (bf16_t)Wq[i];
    wv[i] = (bf16_t)Wv[i];
    wo[i] = (bf16_t)Wo[i];
    const int o = i >> 9, h = i & (HDIM - 1);
    wkT[h * HDIM + o] = (bf16_t)Wk[i];
  }
}

// ---------------------------------------------------------------------------
// Fused kernel: one 64-row tile per workgroup (256 threads = 8 wave32).
// Single __shared__ object => LDS offset 0; all buffers carved from it so
// TDM D# lds_addr offsets are well defined.
// ---------------------------------------------------------------------------
__global__ __launch_bounds__(NTHREADS) void fused_autoattn(
    const float* __restrict__ X,
    const bf16_t* __restrict__ wq, const bf16_t* __restrict__ wkT,
    const bf16_t* __restrict__ wv, const bf16_t* __restrict__ wo,
    const float* __restrict__ bq, const float* __restrict__ bv,
    const float* __restrict__ bo, float* __restrict__ out)
{
  __shared__ char smem[SMEM_TOTAL];
  bf16_t* Xs = (bf16_t*)smem;                    // X tile (bf16)
  bf16_t* Qs = Xs + MTILE * RS;                  // Q, then y
  bf16_t* Gs = Qs + MTILE * RS;                  // G, then ctx

  const int tid  = threadIdx.x;
  const int wave = tid >> 5, lane = tid & 31;
  const size_t row0 = (size_t)blockIdx.x * MTILE;

  char* myslab = smem + SLAB_BASE + wave * 2 * SLAB_BYTES;
  const unsigned myslab_off = (unsigned)(SLAB_BASE + wave * 2 * SLAB_BYTES);

  // ---- Stage 0: load X tile, convert fp32 -> bf16 into LDS --------------
  {
    const float4* src = (const float4*)(X + row0 * HDIM);
    for (int i = tid; i < MTILE * HDIM / 4; i += NTHREADS) {
      const float4 f = src[i];
      const int g = i << 2;
      const int r = g >> 9, c = g & (HDIM - 1);
      bf16_t* p = Xs + r * RS + c;
      p[0] = (bf16_t)f.x; p[1] = (bf16_t)f.y;
      p[2] = (bf16_t)f.z; p[3] = (bf16_t)f.w;
    }
  }
  __syncthreads();

  // ---- Stage 1: Q = X @ Wq^T + bq ---------------------------------------
  gemm_tile(Xs, wq, myslab, myslab_off, Qs, nullptr, bq, wave, lane);
  __syncthreads();

  // ---- Stage 2: G = Q @ Wk  (= Q @ (Wk^T)^T) ----------------------------
  gemm_tile(Qs, wkT, myslab, myslab_off, Gs, nullptr, nullptr, wave, lane);
  __syncthreads();

  // ---- Stage 3: shifted scores, softmax, y (overwrites Qs) --------------
  {
    const int m = tid >> 2, q = tid & 3;         // 4 threads per row
    const bf16_t* xr = Xs + m * RS;
    const bf16_t* gr = Gs + m * RS;

    float sc[NSHIFT] = {0.f, 0.f, 0.f, 0.f, 0.f, 0.f, 0.f, 0.f};
    for (int h = q * 128; h < q * 128 + 128; ++h) {
      const float xv = (float)xr[h];
#pragma unroll
      for (int s = 0; s < NSHIFT; ++s)
        sc[s] += xv * (float)gr[(h + s * PSHIFT) & (HDIM - 1)];
    }
#pragma unroll
    for (int s = 0; s < NSHIFT; ++s) {           // quad reduction (same wave)
      sc[s] += __shfl_xor(sc[s], 1, 32);
      sc[s] += __shfl_xor(sc[s], 2, 32);
    }
    const float scale = 0.04419417382415922f;    // 1/sqrt(512)
    float mx = -3.4e38f;
#pragma unroll
    for (int s = 0; s < NSHIFT; ++s) { sc[s] *= scale; mx = fmaxf(mx, sc[s]); }
    float se = 0.f, at[NSHIFT];
#pragma unroll
    for (int s = 0; s < NSHIFT; ++s) { at[s] = __expf(sc[s] - mx); se += at[s]; }
    const float inv = 1.f / se;
#pragma unroll
    for (int s = 0; s < NSHIFT; ++s) at[s] *= inv;

    bf16_t* yr = Qs + m * RS;                    // Q dead -> y reuses Qs
    for (int h = q * 128; h < q * 128 + 128; ++h) {
      float a = 0.f;
#pragma unroll
      for (int s = 0; s < NSHIFT; ++s)
        a += at[s] * (float)xr[(h - s * PSHIFT) & (HDIM - 1)];
      yr[h] = (bf16_t)a;
    }
  }
  __syncthreads();

  // ---- Stage 4: ctx = y @ Wv^T + bv  (G dead -> write into Gs) ----------
  gemm_tile(Qs, wv, myslab, myslab_off, Gs, nullptr, bv, wave, lane);
  __syncthreads();

  // ---- Stage 5: out = ctx @ Wo^T + bo  -> global fp32 -------------------
  gemm_tile(Gs, wo, myslab, myslab_off, nullptr, out + row0 * HDIM, bo, wave, lane);
}

// ---------------------------------------------------------------------------
// Launch. Inputs: hidden_states, Wq, bq, Wk, bk(unused: softmax-invariant),
// Wv, bv, Wo, bo. Workspace: 4 bf16 weight matrices (2 MB).
// ---------------------------------------------------------------------------
extern "C" void kernel_launch(void* const* d_in, const int* in_sizes, int n_in,
                              void* d_out, int out_size, void* d_ws, size_t ws_size,
                              hipStream_t stream) {
  const float* hs = (const float*)d_in[0];
  const float* Wq = (const float*)d_in[1];
  const float* bq = (const float*)d_in[2];
  const float* Wk = (const float*)d_in[3];
  // d_in[4] = bk: adds the same constant to every score -> cancels in softmax
  const float* Wv = (const float*)d_in[5];
  const float* bv = (const float*)d_in[6];
  const float* Wo = (const float*)d_in[7];
  const float* bo = (const float*)d_in[8];

  bf16_t* wq  = (bf16_t*)d_ws;
  bf16_t* wkT = wq  + HDIM * HDIM;
  bf16_t* wv  = wkT + HDIM * HDIM;
  bf16_t* wo  = wv  + HDIM * HDIM;

  prep_weights<<<512, 256, 0, stream>>>(Wq, Wk, Wv, Wo, wq, wkT, wv, wo);

  fused_autoattn<<<NROWS_TOTAL / MTILE, NTHREADS, 0, stream>>>(
      hs, wq, wkT, wv, wo, bq, bv, bo, (float*)d_out);
}
